// Net_17549236372085
// MI455X (gfx1250) — matrix operand order repaired and checked by reading, not compiled, not run on hardware
//
#include <hip/hip_runtime.h>
#include <hip/hip_bf16.h>
#include <math.h>

typedef __attribute__((ext_vector_type(16))) _Float16 v16h;
typedef __attribute__((ext_vector_type(8)))  _Float16 v8h;
typedef __attribute__((ext_vector_type(8)))  float    v8f;

#define D_FEAT 128
#define HIDDEN 100
#define HPAD   112   // HIDDEN padded to 7 WMMA n-tiles
#define NCLS   10
#define CPAD   16    // NCLS padded to 1 WMMA n-tile

// ---------------- degree / normalization ----------------

__global__ void deg_init_kernel(float* __restrict__ deg, int n) {
  int i = blockIdx.x * blockDim.x + threadIdx.x;
  if (i < n) deg[i] = 1.0f;  // self-loop contributes 1 to every node
}

__global__ void deg_edge_kernel(const int* __restrict__ dst, float* __restrict__ deg, int e) {
  int i = blockIdx.x * blockDim.x + threadIdx.x;
  if (i < e) atomicAdd(&deg[dst[i]], 1.0f);
}

__global__ void dinv_kernel(float* __restrict__ deg, int n) {
  int i = blockIdx.x * blockDim.x + threadIdx.x;
  if (i < n) deg[i] = rsqrtf(deg[i]);  // deg >= 1 always (self-loops)
}

// ---------------- weight pre-padding (f32 -> zero-padded f16) ----------------

__global__ void pad_w1_kernel(const float* __restrict__ W1, _Float16* __restrict__ W1h) {
  const int k = blockIdx.x;      // 0..127
  const int c = threadIdx.x;     // blockDim = 128
  if (c < HPAD)
    W1h[(size_t)k * HPAD + c] =
        (c < HIDDEN) ? (_Float16)W1[(size_t)k * HIDDEN + c] : (_Float16)0.0f;
}

__global__ void pad_w2_kernel(const float* __restrict__ W2, _Float16* __restrict__ W2h) {
  const int i = blockIdx.x * blockDim.x + threadIdx.x;  // 0..2047
  if (i < D_FEAT * CPAD) {
    const int k = i >> 4, c = i & 15;
    W2h[i] = (k < HIDDEN && c < NCLS) ? (_Float16)W2[(size_t)k * NCLS + c]
                                      : (_Float16)0.0f;
  }
}

// ---------------- GEMM1: h1[N,112] = (x[N,128] @ W1h[128,112]) * dinv[row] --------
// One 16x16 output tile per wave; 7 waves per block cover HPAD=112 columns.
// Fragment layouts per CDNA5 ISA 7.12.2. All fragment loads unconditional.

__global__ __launch_bounds__(224)
void gemm1_kernel(const float* __restrict__ x, const _Float16* __restrict__ W1h,
                  const float* __restrict__ dinv, float* __restrict__ h1, int n) {
  const int lane = threadIdx.x & 31;
  const int wave = threadIdx.x >> 5;     // 0..6 -> n-tile
  const int row0 = blockIdx.x * 16;
  const int col0 = wave * 16;
  const int mr   = lane & 15;
  const int hi   = lane >> 4;            // half-wave select

  int arow = row0 + mr; if (arow >= n) arow = n - 1;   // N % 16 == 0 in practice
  const float* xrow = x + (size_t)arow * D_FEAT;

  v8f acc = {};
  #pragma unroll
  for (int kc = 0; kc < D_FEAT; kc += 32) {
    const int abase = kc + 8 * hi;  // A: halves[0..7]=K(abase..+7), [8..15]=K(abase+16..+23)
    v16h a, b;
    #pragma unroll
    for (int j = 0; j < 8; ++j) {
      a[j]     = (_Float16)xrow[abase + j];
      a[j + 8] = (_Float16)xrow[abase + 16 + j];
    }
    const int bn = col0 + (lane & 15);   // B: lane selects N, halves select K (same order as A)
    #pragma unroll
    for (int j = 0; j < 8; ++j) {
      b[j]     = W1h[(size_t)(abase + j) * HPAD + bn];
      b[j + 8] = W1h[(size_t)(abase + 16 + j) * HPAD + bn];
    }
    acc = __builtin_amdgcn_wmma_f32_16x16x32_f16(false, a, false, b, (short)0, acc,
                                                 false, false);
  }
  // C/D layout: VGPR r -> row row0 + 8*hi + r, col = lane&15. Fold dinv[row] into h1.
  const int col = col0 + (lane & 15);
  const bool colok = col < HIDDEN;
  const int rbase = row0 + 8 * hi;
  if (row0 + 16 <= n) {                       // full tile: unconditional epilogue
    const v8f dv = *(const v8f*)(dinv + rbase);   // 32B-aligned contiguous dinv
    #pragma unroll
    for (int r = 0; r < 8; ++r)
      h1[(size_t)(rbase + r) * HPAD + col] = colok ? acc[r] * dv[r] : 0.0f;
  } else {
    #pragma unroll
    for (int r = 0; r < 8; ++r) {
      const int row = rbase + r;
      if (row < n) h1[(size_t)row * HPAD + col] = colok ? acc[r] * dinv[row] : 0.0f;
    }
  }
}

// ---------------- layer-1 aggregation ----------------

// agg1[i] = h1[i] * dinv[i]  (h1 already carries one dinv factor -> self-loop = h1*di^2)
__global__ void selfloop1_kernel(const float* __restrict__ h1, const float* __restrict__ dinv,
                                 float* __restrict__ agg1, int n) {
  const int node = blockIdx.x;
  const int f = threadIdx.x;             // blockDim = 128
  if (f < HIDDEN) {
    const float di = dinv[node];
    agg1[(size_t)node * HPAD + f] = h1[(size_t)node * HPAD + f] * di;
  }
}

// wave-per-edge gather/scale/scatter-add; msg = h1[s] * dinv[d] (dinv[s] pre-folded).
// Hand-unrolled lane striding over the 100-wide row (3 full strides + lane<4 tail).
__global__ __launch_bounds__(256)
void scatter1_kernel(const float* __restrict__ h1, const float* __restrict__ dinv,
                     const int* __restrict__ src, const int* __restrict__ dst,
                     float* __restrict__ agg1, int e) {
  const int w = blockIdx.x * (blockDim.x >> 5) + (threadIdx.x >> 5);  // wave-uniform
  if (w >= e) return;
  const int lane = threadIdx.x & 31;
  __builtin_prefetch(&src[w + 2048], 0, 0);   // stream the edge list (global_prefetch_b8)
  __builtin_prefetch(&dst[w + 2048], 0, 0);
  const int s = src[w];
  const int d = dst[w];
  const float nrm = dinv[d];
  const float* hs = h1 + (size_t)s * HPAD + lane;
  float* ad = agg1 + (size_t)d * HPAD + lane;
  const float m0 = hs[0]  * nrm;
  const float m1 = hs[32] * nrm;
  const float m2 = hs[64] * nrm;
  atomicAdd(&ad[0],  m0);
  atomicAdd(&ad[32], m1);
  atomicAdd(&ad[64], m2);
  if (lane < HIDDEN - 96) {                   // lanes 0..3 cover cols 96..99
    atomicAdd(&ad[96], hs[96] * nrm);
  }
}

// ---------------- bias + ReLU -> zero-padded f16 activations [N,128] ----------------

__global__ void bias_relu_kernel(const float* __restrict__ agg1, const float* __restrict__ b1,
                                 _Float16* __restrict__ h1r, int n) {
  const int node = blockIdx.x;
  const int f = threadIdx.x;             // blockDim = 128, writes all 128 (pad = 0)
  float v = 0.0f;
  if (f < HIDDEN) v = fmaxf(agg1[(size_t)node * HPAD + f] + b1[f], 0.0f);
  h1r[(size_t)node * D_FEAT + f] = (_Float16)v;
}

// ---------------- GEMM2: h2[N,16] = (h1r[N,128] @ W2h[128,16]) * dinv[row] ---------

__global__ __launch_bounds__(256)
void gemm2_kernel(const _Float16* __restrict__ h1r, const _Float16* __restrict__ W2h,
                  const float* __restrict__ dinv, float* __restrict__ h2,
                  float* __restrict__ agg2, int n) {
  const int nt = (n + 15) / 16;
  const int w = blockIdx.x * (blockDim.x >> 5) + (threadIdx.x >> 5);  // wave-uniform
  if (w >= nt) return;
  const int lane = threadIdx.x & 31;
  const int row0 = w * 16;
  const int mr = lane & 15;
  const int hi = lane >> 4;
  int arow = row0 + mr; if (arow >= n) arow = n - 1;
  const _Float16* hrow = h1r + (size_t)arow * D_FEAT;

  v8f acc = {};
  #pragma unroll
  for (int kc = 0; kc < D_FEAT; kc += 32) {
    const int abase = kc + 8 * hi;
    const v8h a0 = *(const v8h*)(hrow + abase);
    const v8h a1 = *(const v8h*)(hrow + abase + 16);
    v16h a, b;
    #pragma unroll
    for (int j = 0; j < 8; ++j) { a[j] = a0[j]; a[j + 8] = a1[j]; }
    const int bn = lane & 15;
    #pragma unroll
    for (int j = 0; j < 8; ++j) {
      b[j]     = W2h[(size_t)(abase + j) * CPAD + bn];
      b[j + 8] = W2h[(size_t)(abase + 16 + j) * CPAD + bn];
    }
    acc = __builtin_amdgcn_wmma_f32_16x16x32_f16(false, a, false, b, (short)0, acc,
                                                 false, false);
  }
  const int col = lane & 15;
  const bool colok = col < NCLS;
  const int rbase = row0 + 8 * hi;
  if (row0 + 16 <= n) {                       // full tile: unconditional epilogue
    const v8f dv = *(const v8f*)(dinv + rbase);
    #pragma unroll
    for (int r = 0; r < 8; ++r) {
      const float hv = (colok ? acc[r] : 0.0f) * dv[r];   // fold dinv[src] into msgs
      h2[(size_t)(rbase + r) * CPAD + col]   = hv;
      agg2[(size_t)(rbase + r) * CPAD + col] = hv * dv[r]; // self-loop = raw*di^2
    }
  } else {
    #pragma unroll
    for (int r = 0; r < 8; ++r) {
      const int row = rbase + r;
      if (row < n) {
        const float di = dinv[row];
        const float hv = (colok ? acc[r] : 0.0f) * di;
        h2[(size_t)row * CPAD + col]   = hv;
        agg2[(size_t)row * CPAD + col] = hv * di;
      }
    }
  }
}

// ---------------- layer-2 aggregation (10-wide): thread per (edge, class) ----------

__global__ void scatter2_kernel(const float* __restrict__ h2, const float* __restrict__ dinv,
                                const int* __restrict__ src, const int* __restrict__ dst,
                                float* __restrict__ agg2, int e) {
  const int gid = blockIdx.x * blockDim.x + threadIdx.x;
  if (gid >= e * NCLS) return;
  const int ed = gid / NCLS;             // mul-shift div by constant
  const int f  = gid - ed * NCLS;
  const int s = src[ed];
  const int d = dst[ed];
  atomicAdd(&agg2[(size_t)d * CPAD + f], h2[(size_t)s * CPAD + f] * dinv[d]);
}

// ---------------- mean pool (column sums) + log_softmax ----------------

__global__ void colsum_kernel(const float* __restrict__ agg2, float* __restrict__ colsum,
                              int n) {
  const int c = blockIdx.x;  // one block per class
  float s = 0.0f;
  for (int i = threadIdx.x; i < n; i += blockDim.x)
    s += agg2[(size_t)i * CPAD + c];
  __shared__ float sm[256];
  sm[threadIdx.x] = s;
  __syncthreads();
  for (int off = 128; off > 0; off >>= 1) {
    if (threadIdx.x < off) sm[threadIdx.x] += sm[threadIdx.x + off];
    __syncthreads();
  }
  if (threadIdx.x == 0) colsum[c] = sm[0];
}

__global__ void finalize_kernel(const float* __restrict__ colsum, const float* __restrict__ b2,
                                float* __restrict__ out, int n) {
  if (threadIdx.x != 0 || blockIdx.x != 0) return;
  float p[NCLS];
  float m = -1e30f;
  const float inv_n = 1.0f / (float)n;
  for (int c = 0; c < NCLS; ++c) {
    p[c] = colsum[c] * inv_n + b2[c];   // mean pool + bias (bias is node-constant)
    m = fmaxf(m, p[c]);
  }
  float s = 0.0f;
  for (int c = 0; c < NCLS; ++c) s += expf(p[c] - m);
  const float lse = logf(s) + m;
  for (int c = 0; c < NCLS; ++c) out[c] = p[c] - lse;
}

// ---------------- launch ----------------

extern "C" void kernel_launch(void* const* d_in, const int* in_sizes, int n_in,
                              void* d_out, int out_size, void* d_ws, size_t ws_size,
                              hipStream_t stream) {
  const float* x  = (const float*)d_in[0];
  const int*   ei = (const int*)d_in[1];     // edge_index [2, E]
  const float* W1 = (const float*)d_in[2];
  const float* b1 = (const float*)d_in[3];
  const float* W2 = (const float*)d_in[4];
  const float* b2 = (const float*)d_in[5];
  float* out = (float*)d_out;

  const int n = in_sizes[0] / D_FEAT;   // 100000
  const int e = in_sizes[1] / 2;        // 1600000
  const int* src = ei;
  const int* dst = ei + e;

  // workspace layout (floats): dinv | h1 | agg1 | W1h | W2h | colsum
  // after scatter1, h1's region is reused for: h1r (f16 N*128) | h2 | agg2
  float* ws = (float*)d_ws;
  size_t off = 0;
  float* dinv = ws + off;  off += (size_t)((n + 127) & ~127u);
  const size_t h1off = off;
  float* h1   = ws + off;  off += (size_t)n * HPAD;
  float* agg1 = ws + off;  off += (size_t)n * HPAD;
  _Float16* W1h = (_Float16*)(ws + off);  off += (D_FEAT * HPAD) / 2;
  _Float16* W2h = (_Float16*)(ws + off);  off += (D_FEAT * CPAD) / 2;
  float* colsum = ws + off; off += 128;
  _Float16* h1r = (_Float16*)(ws + h1off);              // n*128 halves = n*64 floats
  float* h2   = ws + h1off + (size_t)n * (D_FEAT / 2);
  float* agg2 = h2 + (size_t)n * CPAD;
  (void)ws_size; (void)n_in; (void)out_size;            // needs ~90 MB of scratch

  const int T = 256;
  deg_init_kernel <<<(n + T - 1) / T, T, 0, stream>>>(dinv, n);
  deg_edge_kernel <<<(e + T - 1) / T, T, 0, stream>>>(dst, dinv, e);
  dinv_kernel     <<<(n + T - 1) / T, T, 0, stream>>>(dinv, n);

  pad_w1_kernel   <<<D_FEAT, 128, 0, stream>>>(W1, W1h);
  pad_w2_kernel   <<<(D_FEAT * CPAD + T - 1) / T, T, 0, stream>>>(W2, W2h);

  gemm1_kernel    <<<(n + 15) / 16, 224, 0, stream>>>(x, W1h, dinv, h1, n);
  selfloop1_kernel<<<n, 128, 0, stream>>>(h1, dinv, agg1, n);
  scatter1_kernel <<<(e + 7) / 8, 256, 0, stream>>>(h1, dinv, src, dst, agg1, e);

  bias_relu_kernel<<<n, 128, 0, stream>>>(agg1, b1, h1r, n);
  gemm2_kernel    <<<(((n + 15) / 16) + 7) / 8, 256, 0, stream>>>(h1r, W2h, dinv,
                                                                  h2, agg2, n);
  scatter2_kernel <<<(int)(((size_t)e * NCLS + T - 1) / T), T, 0, stream>>>(h2, dinv, src,
                                                                            dst, agg2, e);

  colsum_kernel   <<<NCLS, 256, 0, stream>>>(agg2, colsum, n);
  finalize_kernel <<<1, 32, 0, stream>>>(colsum, b2, out, n);
}
